// GNN_ETM_69441031242462
// MI455X (gfx1250) — matrix operand order repaired and checked by reading, not compiled, not run on hardware
//
#include <hip/hip_runtime.h>

#define NUM_PEAK   28000
#define NUM_GENE   12000
#define N_NODES    40000
#define N_EDGES    640000
#define NUM_TOPICS 100
#define TP         128        /* topics padded to multiple of 64 */
#define GENE_PAD   12032      /* NUM_GENE padded to multiple of 64 */
#define PEAK_PAD   28032      /* NUM_PEAK padded to multiple of 64 */
#define T_HIDDEN   512
#define EMB        64
#define HID        128
#define BATCH      256

typedef __attribute__((ext_vector_type(16))) _Float16 v16h;
typedef __attribute__((ext_vector_type(8)))  _Float16 h8v;
typedef __attribute__((ext_vector_type(8)))  float    v8f;
typedef __fp16 hh2 __attribute__((ext_vector_type(2)));

union V16U { v16h v; h8v h[2]; };

enum { GEMM_NONE = 0, GEMM_RELU = 1, GEMM_LOGDOT = 2 };

// ---------------------------------------------------------------------------
// f16 fragment loads (this code's contiguous k = 16h + s order, same for A and B):
// A 16x32 (MxK): lane<16 row=lane, K=[0..15];
//                lane>=16 row=lane-16, K=[16..31]
// B 32x16 (KxN) from W (NxK row-major): lane<16 col=lane K=[0..15];
//                lane>=16 col=lane-16 K=[16..31]  -> 16 contiguous halves
// ---------------------------------------------------------------------------
__device__ __forceinline__ v16h fragA16(const _Float16* __restrict__ p) {
  V16U u;
  u.h[0] = *(const h8v*)(p);
  u.h[1] = *(const h8v*)(p + 8);
  return u.v;
}
__device__ __forceinline__ v16h fragB16(const _Float16* __restrict__ p) {
  return *(const v16h*)(p);
}

// ---------------------------------------------------------------------------
// C = A(MxK,f16) · B(NpadxK,f16)ᵀ + bias.  Requires M%16==0, Npad%64==0,
// K%32==0 (guaranteed by padding).  One 16x64 strip per wave: one A fragment
// feeds 4 WMMAs per 32-wide K step.  Epilogue guards against Nreal.
// ---------------------------------------------------------------------------
__global__ void gemm_wmma_kernel(const _Float16* __restrict__ A, const _Float16* __restrict__ B,
                                 const float* __restrict__ bias, float* __restrict__ C,
                                 const float* __restrict__ G, float* __restrict__ acc_out,
                                 int M, int N, int Npad, int K, int mode) {
  int lane = threadIdx.x & 31;
  int wid  = threadIdx.x >> 5;
  int NS = Npad >> 6;
  int strip = blockIdx.x * 4 + wid;
  if (strip >= (M >> 4) * NS) return;     // whole-wave uniform exit
  int tm = (strip / NS) << 4;
  int tn = (strip % NS) << 6;
  const int ln = lane & 15, hf = lane >> 4;

  const _Float16* pA  = A + (size_t)(tm + ln) * K + (hf << 4);
  const _Float16* pB0 = B + (size_t)(tn +  0 + ln) * K + (hf << 4);
  const _Float16* pB1 = B + (size_t)(tn + 16 + ln) * K + (hf << 4);
  const _Float16* pB2 = B + (size_t)(tn + 32 + ln) * K + (hf << 4);
  const _Float16* pB3 = B + (size_t)(tn + 48 + ln) * K + (hf << 4);

  v8f acc[4] = {v8f{}, v8f{}, v8f{}, v8f{}};
  for (int k0 = 0; k0 < K; k0 += 32) {
    v16h a = fragA16(pA + k0);
    acc[0] = __builtin_amdgcn_wmma_f32_16x16x32_f16(false, a, false, fragB16(pB0 + k0), (short)0, acc[0], false, false);
    acc[1] = __builtin_amdgcn_wmma_f32_16x16x32_f16(false, a, false, fragB16(pB1 + k0), (short)0, acc[1], false, false);
    acc[2] = __builtin_amdgcn_wmma_f32_16x16x32_f16(false, a, false, fragB16(pB2 + k0), (short)0, acc[2], false, false);
    acc[3] = __builtin_amdgcn_wmma_f32_16x16x32_f16(false, a, false, fragB16(pB3 + k0), (short)0, acc[3], false, false);
  }

  int mb = tm + (hf << 3);
  if (mode == GEMM_LOGDOT) {
    float s = 0.f;
#pragma unroll
    for (int q = 0; q < 4; ++q) {
      int n = tn + (q << 4) + ln;
      if (n < N) {
#pragma unroll
        for (int j = 0; j < 8; ++j)
          s += __logf(acc[q][j] + 1e-6f) * G[(size_t)(mb + j) * N + n];
      }
    }
    for (int o = 16; o > 0; o >>= 1) s += __shfl_down(s, o, 32);
    if (lane == 0) atomicAdd(acc_out, s);
  } else {
#pragma unroll
    for (int q = 0; q < 4; ++q) {
      int n = tn + (q << 4) + ln;
      if (n < N) {
        float bv = (bias != nullptr) ? bias[n] : 0.f;
#pragma unroll
        for (int j = 0; j < 8; ++j) {
          float v = acc[q][j] + bv;
          if (mode == GEMM_RELU) v = fmaxf(v, 0.f);
          C[(size_t)(mb + j) * N + n] = v;
        }
      }
    }
  }
}

// ---------------------------------------------------------------------------
// f32 -> f16 conversion (packed, 8B stores).  n must be a multiple of 4.
// ---------------------------------------------------------------------------
__global__ void cvt_f16_kernel(const float* __restrict__ in, _Float16* __restrict__ out,
                               long long n4) {
  union PK { hh2 h[2]; uint2 u; };
  long long i = (long long)blockIdx.x * blockDim.x + threadIdx.x;
  long long stride = (long long)gridDim.x * blockDim.x;
  for (; i < n4; i += stride) {
    float4 v = ((const float4*)in)[i];
    PK p;
    p.h[0] = __builtin_amdgcn_cvt_pkrtz(v.x, v.y);
    p.h[1] = __builtin_amdgcn_cvt_pkrtz(v.z, v.w);
    ((uint2*)out)[i] = p.u;
  }
}

__global__ void fill_kernel(float* __restrict__ p, float v, long long n) {
  long long i = (long long)blockIdx.x * blockDim.x + threadIdx.x;
  long long stride = (long long)gridDim.x * blockDim.x;
  for (; i < n; i += stride) p[i] = v;
}

__device__ __forceinline__ void atomicMaxF(float* addr, float v) {
  if (v >= 0.f) atomicMax((int*)addr, __float_as_int(v));
  else          atomicMin((unsigned int*)addr, (unsigned int)__float_as_int(v));
}

// BatchNorm over batch axis (rows), in-place. One thread per column.
__global__ void bn_kernel(float* __restrict__ X, const float* __restrict__ gamma,
                          const float* __restrict__ beta, int rows, int F) {
  int c = blockIdx.x * blockDim.x + threadIdx.x;
  if (c >= F) return;
  float s = 0.f, sq = 0.f;
  for (int r = 0; r < rows; ++r) {
    float x = X[(size_t)r * F + c];
    s += x; sq += x * x;
  }
  float m = s / rows;
  float var = sq / rows - m * m;
  float sc = rsqrtf(var + 1e-5f) * gamma[c];
  float bb = beta[c];
  for (int r = 0; r < rows; ++r)
    X[(size_t)r * F + c] = (X[(size_t)r * F + c] - m) * sc + bb;
}

// GATv2 edge pass 1: logit + segment-max(dst)
__global__ void edge_logit_kernel(const int* __restrict__ ei, int nE, int nNodes,
                                  const float* __restrict__ xl, const float* __restrict__ xr,
                                  const float* __restrict__ att,
                                  float* __restrict__ logit, float* __restrict__ mmax, int H) {
  int e = blockIdx.x * blockDim.x + threadIdx.x;
  if (e >= nE + nNodes) return;
  int s, d;
  if (e < nE) { s = ei[e]; d = ei[nE + e]; } else { s = d = e - nE; }
  const float* pl = xl + (size_t)s * H;
  const float* pr = xr + (size_t)d * H;
  float acc = 0.f;
  for (int h = 0; h < H; h += 4) {
    float4 a = *(const float4*)(pl + h);
    float4 b = *(const float4*)(pr + h);
    float4 t = *(const float4*)(att + h);
    float e0 = a.x + b.x; e0 = e0 > 0.f ? e0 : 0.2f * e0;
    float e1 = a.y + b.y; e1 = e1 > 0.f ? e1 : 0.2f * e1;
    float e2 = a.z + b.z; e2 = e2 > 0.f ? e2 : 0.2f * e2;
    float e3 = a.w + b.w; e3 = e3 > 0.f ? e3 : 0.2f * e3;
    acc += t.x * e0 + t.y * e1 + t.z * e2 + t.w * e3;
  }
  logit[e] = acc;
  atomicMaxF(&mmax[d], acc);
}

// pass 2: ex = exp(logit - m[dst]); segment-sum(dst)
__global__ void edge_exp_kernel(const int* __restrict__ ei, int nE, int nNodes,
                                float* __restrict__ logit, const float* __restrict__ mmax,
                                float* __restrict__ den) {
  int e = blockIdx.x * blockDim.x + threadIdx.x;
  if (e >= nE + nNodes) return;
  int d = (e < nE) ? ei[nE + e] : (e - nE);
  float ex = __expf(logit[e] - mmax[d]);
  logit[e] = ex;
  atomicAdd(&den[d], ex);
}

// pass 3: out[dst] += alpha * xl[src]
__global__ void edge_accum_kernel(const int* __restrict__ ei, int nE, int nNodes,
                                  const float* __restrict__ ex, const float* __restrict__ den,
                                  const float* __restrict__ xl, float* __restrict__ out, int H) {
  int e = blockIdx.x * blockDim.x + threadIdx.x;
  if (e >= nE + nNodes) return;
  int s, d;
  if (e < nE) { s = ei[e]; d = ei[nE + e]; } else { s = d = e - nE; }
  float alpha = ex[e] / (den[d] + 1e-16f);
  const float* pl = xl + (size_t)s * H;
  float* po = out + (size_t)d * H;
  for (int h = 0; h < H; ++h) atomicAdd(&po[h], alpha * pl[h]);
}

__global__ void bias_relu_kernel(float* __restrict__ x, const float* __restrict__ bias,
                                 long long n, int Hmask) {
  long long i = (long long)blockIdx.x * blockDim.x + threadIdx.x;
  long long stride = (long long)gridDim.x * blockDim.x;
  for (; i < n; i += stride) x[i] = fmaxf(x[i] + bias[(int)i & Hmask], 0.f);
}

// row softmax over cols_real, zero the padded tail; one block per row
__global__ void rowsoftmax_kernel(const float* __restrict__ X, float* __restrict__ Y,
                                  int cols_real, int cols_pad) {
  __shared__ float red[128];
  int r = blockIdx.x, tid = threadIdx.x;
  const float* x = X + (size_t)r * cols_pad;
  float* y = Y + (size_t)r * cols_pad;
  float mx = -3.4e38f;
  for (int c = tid; c < cols_real; c += 128) mx = fmaxf(mx, x[c]);
  red[tid] = mx; __syncthreads();
  for (int s = 64; s > 0; s >>= 1) { if (tid < s) red[tid] = fmaxf(red[tid], red[tid + s]); __syncthreads(); }
  mx = red[0]; __syncthreads();
  float sum = 0.f;
  for (int c = tid; c < cols_real; c += 128) sum += __expf(x[c] - mx);
  red[tid] = sum; __syncthreads();
  for (int s = 64; s > 0; s >>= 1) { if (tid < s) red[tid] += red[tid + s]; __syncthreads(); }
  sum = red[0];
  for (int c = tid; c < cols_pad; c += 128)
    y[c] = (c < cols_real) ? (__expf(x[c] - mx) / sum) : 0.f;
}

// softmax along columns (axis=0) with row stride ld, in-place; one block/col
__global__ void colsoftmax_kernel(float* __restrict__ S, int R, int ld) {
  __shared__ float red[256];
  int c = blockIdx.x, tid = threadIdx.x;
  float mx = -3.4e38f;
  for (int r = tid; r < R; r += 256) mx = fmaxf(mx, S[(size_t)r * ld + c]);
  red[tid] = mx; __syncthreads();
  for (int s = 128; s > 0; s >>= 1) { if (tid < s) red[tid] = fmaxf(red[tid], red[tid + s]); __syncthreads(); }
  mx = red[0]; __syncthreads();
  float sum = 0.f;
  for (int r = tid; r < R; r += 256) sum += __expf(S[(size_t)r * ld + c] - mx);
  red[tid] = sum; __syncthreads();
  for (int s = 128; s > 0; s >>= 1) { if (tid < s) red[tid] += red[tid + s]; __syncthreads(); }
  sum = red[0];
  for (int r = tid; r < R; r += 256)
    S[(size_t)r * ld + c] = __expf(S[(size_t)r * ld + c] - mx) / sum;
}

// KL: -0.5/B * sum_b sum_t (1 + 2l - mu^2 - exp(2l)); single block of 256
__global__ void kl_kernel(const float* __restrict__ mu, const float* __restrict__ lsr,
                          int T, int ld, float* __restrict__ out) {
  __shared__ float red[256];
  int b = threadIdx.x;
  float s = 0.f;
  for (int t = 0; t < T; ++t) {
    float m = mu[(size_t)b * ld + t];
    float l = 2.f * lsr[(size_t)b * ld + t];
    s += 1.f + l - m * m - __expf(l);
  }
  red[b] = s; __syncthreads();
  for (int st = 128; st > 0; st >>= 1) { if (b < st) red[b] += red[b + st]; __syncthreads(); }
  if (b == 0) out[0] = -0.5f * red[0] / (float)BATCH;
}

__global__ void combine_kernel(const float* accG, const float* accP,
                               const float* klG, const float* klP, float* out) {
  if (threadIdx.x == 0) {
    out[0] = -(accG[0] + accP[0]) / (float)BATCH;
    out[1] = klG[0] + klP[0];
  }
}

// ---------------------------------------------------------------------------
// Host-side orchestration
// ---------------------------------------------------------------------------
static void gemm(hipStream_t st, const _Float16* A, const _Float16* B, const float* bias,
                 float* C, const float* G, float* acc, int M, int N, int Npad, int K, int mode) {
  int strips = (M / 16) * (Npad / 64);
  gemm_wmma_kernel<<<(strips + 3) / 4, 128, 0, st>>>(A, B, bias, C, G, acc, M, N, Npad, K, mode);
}

static void cvt(hipStream_t st, const float* in, _Float16* out, long long n) {
  long long n4 = n / 4;
  int blocks = (int)((n4 + 255) / 256); if (blocks > 4096) blocks = 4096;
  cvt_f16_kernel<<<blocks, 256, 0, st>>>(in, out, n4);
}

static void fillf(hipStream_t st, float* p, float v, long long n) {
  int blocks = (int)((n + 255) / 256); if (blocks > 4096) blocks = 4096;
  fill_kernel<<<blocks, 256, 0, st>>>(p, v, n);
}

extern "C" void kernel_launch(void* const* d_in, const int* in_sizes, int n_in,
                              void* d_out, int out_size, void* d_ws, size_t ws_size,
                              hipStream_t stream) {
  const float* Gene     = (const float*)d_in[0];
  const float* Gene_n   = (const float*)d_in[1];
  const float* Peak     = (const float*)d_in[2];
  const float* Peak_n   = (const float*)d_in[3];
  const float* fm       = (const float*)d_in[4];
  const float* Wl1      = (const float*)d_in[5];
  const float* bl1      = (const float*)d_in[6];
  const float* Wr1      = (const float*)d_in[7];
  const float* br1      = (const float*)d_in[8];
  const float* att1     = (const float*)d_in[9];
  const float* bias1    = (const float*)d_in[10];
  const float* Wl2      = (const float*)d_in[11];
  const float* bl2      = (const float*)d_in[12];
  const float* Wr2      = (const float*)d_in[13];
  const float* br2      = (const float*)d_in[14];
  const float* att2     = (const float*)d_in[15];
  const float* bias2    = (const float*)d_in[16];
  const float* gW1      = (const float*)d_in[17];
  const float* gb1      = (const float*)d_in[18];
  const float* gg1      = (const float*)d_in[19];
  const float* gbe1     = (const float*)d_in[20];
  const float* gW2      = (const float*)d_in[21];
  const float* gb2      = (const float*)d_in[22];
  const float* gg2      = (const float*)d_in[23];
  const float* gbe2     = (const float*)d_in[24];
  const float* pW1      = (const float*)d_in[25];
  const float* pb1      = (const float*)d_in[26];
  const float* pg1      = (const float*)d_in[27];
  const float* pbe1     = (const float*)d_in[28];
  const float* pW2      = (const float*)d_in[29];
  const float* pb2      = (const float*)d_in[30];
  const float* pg2      = (const float*)d_in[31];
  const float* pbe2     = (const float*)d_in[32];
  const float* mu_W     = (const float*)d_in[33];
  const float* mu_b     = (const float*)d_in[34];
  const float* ls_W     = (const float*)d_in[35];
  const float* ls_b     = (const float*)d_in[36];
  const float* alphas_W = (const float*)d_in[37];
  const float* alphas_sW= (const float*)d_in[38];
  const int*   ei       = (const int*)d_in[39];
  float* out = (float*)d_out;
  (void)ws_size; (void)n_in; (void)in_sizes; (void)out_size;

  // ---- workspace layout ----
  char* base = (char*)d_ws;
  size_t off = 0;
  auto allocB = [&](size_t bytes) { char* p = base + off; off = (off + bytes + 15) & ~(size_t)15; return p; };
  auto aF = [&](size_t n) { return (float*)allocB(n * 4); };
  auto aH = [&](size_t n) { return (_Float16*)allocB(n * 2); };

  // f32 buffers
  float* t1      = aF(BATCH * T_HIDDEN);
  float* t2      = aF(BATCH * T_HIDDEN);
  float* mu_d    = aF(BATCH * TP);
  float* lsr_d   = aF(BATCH * TP);
  float* mu_j    = aF(BATCH * TP);
  float* lsr_j   = aF(BATCH * TP);
  float* theta   = aF(BATCH * TP);          // reused for d then j
  float* xl1     = aF((size_t)N_NODES * HID);  // reused: xl2, then Sg f32
  float* xr1     = aF((size_t)N_NODES * HID);  // reused: xr2, then Sp f32
  float* gat1    = aF((size_t)N_NODES * HID);  // becomes h
  float* gat2    = aF((size_t)N_NODES * EMB);  // becomes emb
  float* logit   = aF(N_EDGES + N_NODES);
  float* mmax    = aF(N_NODES);
  float* den     = aF(N_NODES);
  float* mu_b_p  = aF(TP);
  float* ls_b_p  = aF(TP);
  float* accG    = aF(4);  // [0]=accG [1]=accP [2]=klG [3]=klP
  float* accP = accG + 1; float* klG = accG + 2; float* klP = accG + 3;

  // f16 staging buffers
  _Float16* Gh    = aH((size_t)BATCH * NUM_GENE);
  _Float16* Ph    = aH((size_t)BATCH * NUM_PEAK);
  _Float16* gW1h  = aH((size_t)T_HIDDEN * NUM_GENE);
  _Float16* pW1h  = aH((size_t)T_HIDDEN * NUM_PEAK);
  _Float16* gW2h  = aH(T_HIDDEN * T_HIDDEN);
  _Float16* pW2h  = aH(T_HIDDEN * T_HIDDEN);
  _Float16* t1h   = aH(BATCH * T_HIDDEN);
  _Float16* t2h   = aH(BATCH * T_HIDDEN);
  _Float16* muWh  = aH(TP * T_HIDDEN);      // zero-padded rows 100..127
  _Float16* lsWh  = aH(TP * T_HIDDEN);
  _Float16* fmh   = aH((size_t)N_NODES * EMB);
  _Float16* Wl1h  = aH(HID * EMB);
  _Float16* Wr1h  = aH(HID * EMB);
  _Float16* Wl2h  = aH(EMB * HID);
  _Float16* Wr2h  = aH(EMB * HID);
  _Float16* aWh   = aH(TP * EMB);           // zero-padded
  _Float16* aSWh  = aH(TP * EMB);
  _Float16* hh    = aH((size_t)N_NODES * HID);
  _Float16* embh  = aH((size_t)N_NODES * EMB);
  _Float16* Sgh   = aH((size_t)GENE_PAD * TP);  // zero tail rows
  _Float16* Sph   = aH((size_t)PEAK_PAD * TP);
  _Float16* thdh  = aH(BATCH * TP);
  _Float16* thjh  = aH(BATCH * TP);

  const int nEtot = N_EDGES + N_NODES;
  const int eblk  = (nEtot + 255) / 256;

  // ---- one-time stagings: weights -> f16 (zero-pad where needed) ----
  cvt(stream, gW1, gW1h, (long long)T_HIDDEN * NUM_GENE);
  cvt(stream, pW1, pW1h, (long long)T_HIDDEN * NUM_PEAK);
  cvt(stream, gW2, gW2h, T_HIDDEN * T_HIDDEN);
  cvt(stream, pW2, pW2h, T_HIDDEN * T_HIDDEN);
  fillf(stream, (float*)muWh, 0.f, TP * T_HIDDEN / 2);
  fillf(stream, (float*)lsWh, 0.f, TP * T_HIDDEN / 2);
  cvt(stream, mu_W, muWh, NUM_TOPICS * T_HIDDEN);
  cvt(stream, ls_W, lsWh, NUM_TOPICS * T_HIDDEN);
  cvt(stream, Wl1, Wl1h, HID * EMB);
  cvt(stream, Wr1, Wr1h, HID * EMB);
  cvt(stream, Wl2, Wl2h, EMB * HID);
  cvt(stream, Wr2, Wr2h, EMB * HID);
  fillf(stream, (float*)aWh,  0.f, TP * EMB / 2);
  fillf(stream, (float*)aSWh, 0.f, TP * EMB / 2);
  cvt(stream, alphas_W,  aWh,  NUM_TOPICS * EMB);
  cvt(stream, alphas_sW, aSWh, NUM_TOPICS * EMB);
  fillf(stream, mu_b_p, 0.f, TP);
  fillf(stream, ls_b_p, 0.f, TP);
  hipMemcpyAsync(mu_b_p, mu_b, NUM_TOPICS * sizeof(float), hipMemcpyDeviceToDevice, stream);
  hipMemcpyAsync(ls_b_p, ls_b, NUM_TOPICS * sizeof(float), hipMemcpyDeviceToDevice, stream);
  cvt(stream, Gene_n, Gh, (long long)BATCH * NUM_GENE);
  cvt(stream, Peak_n, Ph, (long long)BATCH * NUM_PEAK);
  cvt(stream, fm, fmh, (long long)N_NODES * EMB);

  // ---- gene encoder ----
  gemm(stream, Gh, gW1h, gb1, t1, nullptr, nullptr, BATCH, T_HIDDEN, T_HIDDEN, NUM_GENE, GEMM_RELU);
  bn_kernel<<<2, 256, 0, stream>>>(t1, gg1, gbe1, BATCH, T_HIDDEN);
  cvt(stream, t1, t1h, BATCH * T_HIDDEN);
  gemm(stream, t1h, gW2h, gb2, t2, nullptr, nullptr, BATCH, T_HIDDEN, T_HIDDEN, T_HIDDEN, GEMM_RELU);
  bn_kernel<<<2, 256, 0, stream>>>(t2, gg2, gbe2, BATCH, T_HIDDEN);
  cvt(stream, t2, t2h, BATCH * T_HIDDEN);
  gemm(stream, t2h, muWh, mu_b_p, mu_d,  nullptr, nullptr, BATCH, TP, TP, T_HIDDEN, GEMM_NONE);
  gemm(stream, t2h, lsWh, ls_b_p, lsr_d, nullptr, nullptr, BATCH, TP, TP, T_HIDDEN, GEMM_NONE);
  rowsoftmax_kernel<<<BATCH, 128, 0, stream>>>(mu_d, theta, NUM_TOPICS, TP);
  cvt(stream, theta, thdh, BATCH * TP);
  kl_kernel<<<1, 256, 0, stream>>>(mu_d, lsr_d, NUM_TOPICS, TP, klG);

  // ---- peak encoder ----
  gemm(stream, Ph, pW1h, pb1, t1, nullptr, nullptr, BATCH, T_HIDDEN, T_HIDDEN, NUM_PEAK, GEMM_RELU);
  bn_kernel<<<2, 256, 0, stream>>>(t1, pg1, pbe1, BATCH, T_HIDDEN);
  cvt(stream, t1, t1h, BATCH * T_HIDDEN);
  gemm(stream, t1h, pW2h, pb2, t2, nullptr, nullptr, BATCH, T_HIDDEN, T_HIDDEN, T_HIDDEN, GEMM_RELU);
  bn_kernel<<<2, 256, 0, stream>>>(t2, pg2, pbe2, BATCH, T_HIDDEN);
  cvt(stream, t2, t2h, BATCH * T_HIDDEN);
  gemm(stream, t2h, muWh, mu_b_p, mu_j,  nullptr, nullptr, BATCH, TP, TP, T_HIDDEN, GEMM_NONE);
  gemm(stream, t2h, lsWh, ls_b_p, lsr_j, nullptr, nullptr, BATCH, TP, TP, T_HIDDEN, GEMM_NONE);
  rowsoftmax_kernel<<<BATCH, 128, 0, stream>>>(mu_j, theta, NUM_TOPICS, TP);
  cvt(stream, theta, thjh, BATCH * TP);
  kl_kernel<<<1, 256, 0, stream>>>(mu_j, lsr_j, NUM_TOPICS, TP, klP);

  // ---- GATv2 layer 1 (64 -> 128) ----
  gemm(stream, fmh, Wl1h, bl1, xl1, nullptr, nullptr, N_NODES, HID, HID, EMB, GEMM_NONE);
  gemm(stream, fmh, Wr1h, br1, xr1, nullptr, nullptr, N_NODES, HID, HID, EMB, GEMM_NONE);
  fillf(stream, mmax, -3.4e38f, N_NODES);
  fillf(stream, den, 0.f, N_NODES);
  fillf(stream, gat1, 0.f, (long long)N_NODES * HID);
  edge_logit_kernel<<<eblk, 256, 0, stream>>>(ei, N_EDGES, N_NODES, xl1, xr1, att1, logit, mmax, HID);
  edge_exp_kernel<<<eblk, 256, 0, stream>>>(ei, N_EDGES, N_NODES, logit, mmax, den);
  edge_accum_kernel<<<eblk, 256, 0, stream>>>(ei, N_EDGES, N_NODES, logit, den, xl1, gat1, HID);
  bias_relu_kernel<<<4096, 256, 0, stream>>>(gat1, bias1, (long long)N_NODES * HID, HID - 1);
  cvt(stream, gat1, hh, (long long)N_NODES * HID);

  // ---- GATv2 layer 2 (128 -> 64); xl2/xr2 reuse xl1/xr1 ----
  gemm(stream, hh, Wl2h, bl2, xl1, nullptr, nullptr, N_NODES, EMB, EMB, HID, GEMM_NONE);
  gemm(stream, hh, Wr2h, br2, xr1, nullptr, nullptr, N_NODES, EMB, EMB, HID, GEMM_NONE);
  fillf(stream, mmax, -3.4e38f, N_NODES);
  fillf(stream, den, 0.f, N_NODES);
  fillf(stream, gat2, 0.f, (long long)N_NODES * EMB);
  edge_logit_kernel<<<eblk, 256, 0, stream>>>(ei, N_EDGES, N_NODES, xl1, xr1, att2, logit, mmax, EMB);
  edge_exp_kernel<<<eblk, 256, 0, stream>>>(ei, N_EDGES, N_NODES, logit, mmax, den);
  edge_accum_kernel<<<eblk, 256, 0, stream>>>(ei, N_EDGES, N_NODES, logit, den, xl1, gat2, EMB);
  bias_relu_kernel<<<4096, 256, 0, stream>>>(gat2, bias2, (long long)N_NODES * EMB, EMB - 1);
  cvt(stream, gat2, embh, (long long)N_NODES * EMB);

  // ---- topic mixing: Sg = softmax0(rho @ aWᵀ), Sp = softmax0(eta @ aSWᵀ) ----
  float* Sg = xl1;  // NUM_GENE x TP (f32)
  float* Sp = xr1;  // NUM_PEAK x TP (f32)
  const _Float16* etah = embh;                           // emb[:NUM_PEAK]
  const _Float16* rhoh = embh + (size_t)NUM_PEAK * EMB;  // emb[NUM_PEAK:]
  gemm(stream, rhoh, aWh,  nullptr, Sg, nullptr, nullptr, NUM_GENE, TP, TP, EMB, GEMM_NONE);
  gemm(stream, etah, aSWh, nullptr, Sp, nullptr, nullptr, NUM_PEAK, TP, TP, EMB, GEMM_NONE);
  colsoftmax_kernel<<<NUM_TOPICS, 256, 0, stream>>>(Sg, NUM_GENE, TP);
  colsoftmax_kernel<<<NUM_TOPICS, 256, 0, stream>>>(Sp, NUM_PEAK, TP);
  fillf(stream, (float*)Sgh, 0.f, (long long)GENE_PAD * TP / 2);
  fillf(stream, (float*)Sph, 0.f, (long long)PEAK_PAD * TP / 2);
  cvt(stream, Sg, Sgh, (long long)NUM_GENE * TP);
  cvt(stream, Sp, Sph, (long long)NUM_PEAK * TP);

  // ---- fused recon: sum(log(theta @ Ssmᵀ + 1e-6) * X) ----
  fillf(stream, accG, 0.f, 2);
  gemm(stream, thdh, Sgh, nullptr, nullptr, Gene, accG, BATCH, NUM_GENE, GENE_PAD, TP, GEMM_LOGDOT);
  gemm(stream, thjh, Sph, nullptr, nullptr, Peak, accP, BATCH, NUM_PEAK, PEAK_PAD, TP, GEMM_LOGDOT);

  combine_kernel<<<1, 32, 0, stream>>>(accG, accP, klG, klP, out);
}
